// AttentionLayer_2559800508633
// MI455X (gfx1250) — compile-verified, matrix operand order checked by reading, not run on hardware
//
#include <hip/hip_runtime.h>

// ---------------------------------------------------------------------------
// Cross-attention layer for MI455X (gfx1250, wave32, WMMA).
// Outputs: [output (B,S1,1024) f32] ++ [attn_weights (B,H,S1,S2) f32]
// Strategy: f32 -> bf16 once; all GEMMs via v_wmma_f32_16x16x32_bf16;
// scores written to d_out attn region, softmax in place, PV reads it back.
// All dims compile-time so epilogue addressing is shifts/masks (no int div).
// ---------------------------------------------------------------------------

typedef __bf16 bf16_t;
typedef __attribute__((ext_vector_type(16))) __bf16 v16bf;
typedef __attribute__((ext_vector_type(8)))  __bf16 v8bf;
typedef __attribute__((ext_vector_type(8)))  float  v8f;

constexpr int NHEADS = 16;
constexpr int DHEAD  = 64;
constexpr int DEMB   = 1024;
constexpr int BATCH  = 2;
constexpr int SEQ    = 2048;             // S1 == S2
constexpr int MROWS  = BATCH * SEQ;      // 4096

// --- WMMA helpers -----------------------------------------------------------

static __device__ inline v8f wmma_bf16(v16bf a, v16bf b, v8f c) {
  // (neg_a, A, neg_b, B, c_mod, C, reuse_a, reuse_b)
  return __builtin_amdgcn_wmma_f32_16x16x32_bf16(false, a, false, b,
                                                 (short)0, c, false, false);
}

// A/B operand loader. `p` points at element (row, k0+kb) of a row-major
// matrix whose k-dimension is contiguous. Chunk layout per ISA 7.12.2:
// v16bf elems 0..7  = K kb..kb+7   (VGPR 0..3)
// v16bf elems 8..15 = K 16+kb..+7  (VGPR 4..7)
static __device__ inline v16bf load_bf16_operand(const bf16_t* __restrict__ p) {
  v8bf lo = *(const v8bf*)(p);        // 16B -> global_load_b128
  v8bf hi = *(const v8bf*)(p + 16);
  v16bf r;
#pragma unroll
  for (int i = 0; i < 8; ++i) { r[i] = lo[i]; r[8 + i] = hi[i]; }
  return r;
}

// Same layout, but source is f32 (softmaxed attention weights) converted
// to bf16 in registers.
static __device__ inline v16bf load_f32_operand_as_bf16(const float* __restrict__ p) {
  v8f lo = *(const v8f*)(p);
  v8f hi = *(const v8f*)(p + 16);
  v16bf r;
#pragma unroll
  for (int i = 0; i < 8; ++i) { r[i] = (bf16_t)lo[i]; r[8 + i] = (bf16_t)hi[i]; }
  return r;
}

// --- Kernel 0: f32 -> bf16 conversion ---------------------------------------

__global__ void cvt_f32_to_bf16(const float* __restrict__ src,
                                bf16_t* __restrict__ dst, int n) {
  int i = blockIdx.x * blockDim.x + threadIdx.x;
  if (i < n) dst[i] = (bf16_t)src[i];
}

// --- Kernel 1: Y = X @ W^T + bias (projections) ------------------------------
// X: [MROWS, DEMB] bf16 row-major.  W: [DEMB, DEMB] bf16 row-major (torch).
// MODE 0: outF f32 [MROWS, DEMB]                   (final output projection)
// MODE 1: outB bf16 head-split [B, H, SEQ, 64]     (Q, K)
// MODE 2: outB bf16 head-split-T [B, H, 64, SEQ]   (V, pre-transposed for PV)
template <int MODE>
__global__ __launch_bounds__(128) void gemm_xwT_wmma(
    const bf16_t* __restrict__ X, const bf16_t* __restrict__ W,
    const float* __restrict__ bias, float* __restrict__ outF,
    bf16_t* __restrict__ outB) {
  const int lane = threadIdx.x & 31;
  const int wave = threadIdx.x >> 5;
  const int ln   = lane & 15;
  const int kb   = (lane >> 4) << 3;        // 0 or 8
  const int m0   = blockIdx.x * 128 + (wave >> 1) * 64;
  const int n0   = blockIdx.y * 128 + (wave & 1) * 64;

  v8f acc[4][4];
#pragma unroll
  for (int i = 0; i < 4; ++i)
#pragma unroll
    for (int j = 0; j < 4; ++j) acc[i][j] = (v8f)(0.0f);

  for (int k0 = 0; k0 < DEMB; k0 += 32) {
    v16bf a[4], b[4];
#pragma unroll
    for (int i = 0; i < 4; ++i) {
      const bf16_t* pa = X + (size_t)(m0 + 16 * i + ln) * DEMB + k0 + kb;
      __builtin_prefetch(pa + 64, 0, 3);    // keep near-cache (WGP) resident
      a[i] = load_bf16_operand(pa);
    }
#pragma unroll
    for (int j = 0; j < 4; ++j)
      b[j] = load_bf16_operand(W + (size_t)(n0 + 16 * j + ln) * DEMB + k0 + kb);
#pragma unroll
    for (int i = 0; i < 4; ++i)
#pragma unroll
      for (int j = 0; j < 4; ++j) acc[i][j] = wmma_bf16(a[i], b[j], acc[i][j]);
  }

  const int mrow = (lane >> 4) << 3;        // 8*(lane/16)
#pragma unroll
  for (int i = 0; i < 4; ++i)
#pragma unroll
    for (int j = 0; j < 4; ++j)
#pragma unroll
      for (int r = 0; r < 8; ++r) {
        const int m = m0 + 16 * i + r + mrow;
        const int n = n0 + 16 * j + ln;
        const float v = acc[i][j][r] + bias[n];
        if constexpr (MODE == 0) {
          outF[(size_t)m * DEMB + n] = v;
        } else {
          const int bb = m >> 11, s = m & (SEQ - 1);   // SEQ == 2048
          const int h = n >> 6, d = n & 63;
          if constexpr (MODE == 1)
            outB[(((size_t)bb * NHEADS + h) * SEQ + s) * DHEAD + d] = (bf16_t)v;
          else
            outB[(((size_t)bb * NHEADS + h) * DHEAD + d) * (size_t)SEQ + s] = (bf16_t)v;
        }
      }
}

// --- Kernel 2: scores = (Q K^T) * 1/sqrt(64) --------------------------------
// Q: [BH, SEQ, 64] bf16.  Km: [BH, SEQ, 64] bf16.  scores: [BH, SEQ, SEQ] f32.
__global__ __launch_bounds__(128) void scores_wmma(
    const bf16_t* __restrict__ Q, const bf16_t* __restrict__ Km,
    float* __restrict__ scores) {
  const int bh   = blockIdx.z;
  const int lane = threadIdx.x & 31;
  const int wave = threadIdx.x >> 5;
  const int ln   = lane & 15;
  const int kb   = (lane >> 4) << 3;
  const int q0   = blockIdx.x * 128 + (wave >> 1) * 64;
  const int j0   = blockIdx.y * 128 + (wave & 1) * 64;
  const bf16_t* Qb = Q + (size_t)bh * SEQ * DHEAD;
  const bf16_t* Kb = Km + (size_t)bh * SEQ * DHEAD;

  v8f acc[4][4];
#pragma unroll
  for (int i = 0; i < 4; ++i)
#pragma unroll
    for (int j = 0; j < 4; ++j) acc[i][j] = (v8f)(0.0f);

#pragma unroll
  for (int k0 = 0; k0 < DHEAD; k0 += 32) {
    v16bf a[4], b[4];
#pragma unroll
    for (int i = 0; i < 4; ++i)
      a[i] = load_bf16_operand(Qb + (size_t)(q0 + 16 * i + ln) * DHEAD + k0 + kb);
#pragma unroll
    for (int j = 0; j < 4; ++j)
      b[j] = load_bf16_operand(Kb + (size_t)(j0 + 16 * j + ln) * DHEAD + k0 + kb);
#pragma unroll
    for (int i = 0; i < 4; ++i)
#pragma unroll
      for (int j = 0; j < 4; ++j) acc[i][j] = wmma_bf16(a[i], b[j], acc[i][j]);
  }

  const int mrow = (lane >> 4) << 3;
  float* out = scores + (size_t)bh * SEQ * SEQ;
#pragma unroll
  for (int i = 0; i < 4; ++i)
#pragma unroll
    for (int j = 0; j < 4; ++j)
#pragma unroll
      for (int r = 0; r < 8; ++r) {
        const int q = q0 + 16 * i + r + mrow;
        const int jj = j0 + 16 * j + ln;
        out[(size_t)q * SEQ + jj] = acc[i][j][r] * 0.125f;  // 1/sqrt(64)
      }
}

// --- Kernel 3: row softmax in place -----------------------------------------
__global__ __launch_bounds__(256) void softmax_rows(float* __restrict__ w) {
  __shared__ float red[256];
  float* p = w + (size_t)blockIdx.x * SEQ;
  const int t = threadIdx.x;

  float mx = -3.402823466e38f;
#pragma unroll
  for (int i = t; i < SEQ; i += 256) mx = fmaxf(mx, p[i]);
  red[t] = mx; __syncthreads();
#pragma unroll
  for (int s = 128; s > 0; s >>= 1) {
    if (t < s) red[t] = fmaxf(red[t], red[t + s]);
    __syncthreads();
  }
  mx = red[0]; __syncthreads();

  float sum = 0.0f;
#pragma unroll
  for (int i = t; i < SEQ; i += 256) {
    float e = __expf(p[i] - mx);
    p[i] = e;
    sum += e;
  }
  red[t] = sum; __syncthreads();
#pragma unroll
  for (int s = 128; s > 0; s >>= 1) {
    if (t < s) red[t] += red[t + s];
    __syncthreads();
  }
  const float inv = 1.0f / red[0];
#pragma unroll
  for (int i = t; i < SEQ; i += 256) p[i] *= inv;
}

// --- Kernel 4: attn_out = P @ V ---------------------------------------------
// P: [BH, SEQ, SEQ] f32 (softmaxed).  Vt: [BH, 64, SEQ] bf16 (pre-transposed).
// AO: [B, SEQ, 1024] bf16, written interleaved back to embedding layout.
__global__ __launch_bounds__(128) void av_wmma(
    const float* __restrict__ P, const bf16_t* __restrict__ Vt,
    bf16_t* __restrict__ AO) {
  const int bh = blockIdx.z;
  const int b  = bh >> 4, h = bh & (NHEADS - 1);
  const int lane = threadIdx.x & 31;
  const int wave = threadIdx.x >> 5;
  const int ln   = lane & 15;
  const int kb   = (lane >> 4) << 3;
  const int q0   = blockIdx.x * 256 + wave * 64;  // 4 waves stacked on M
  const float*  Pb = P + (size_t)bh * SEQ * SEQ;
  const bf16_t* Vb = Vt + (size_t)bh * DHEAD * SEQ;

  v8f acc[4][4];
#pragma unroll
  for (int i = 0; i < 4; ++i)
#pragma unroll
    for (int j = 0; j < 4; ++j) acc[i][j] = (v8f)(0.0f);

  for (int k0 = 0; k0 < SEQ; k0 += 32) {
    v16bf a[4], bmat[4];
#pragma unroll
    for (int i = 0; i < 4; ++i) {
      const float* pa = Pb + (size_t)(q0 + 16 * i + ln) * SEQ + k0 + kb;
      __builtin_prefetch(pa + 128, 0, 3);
      a[i] = load_f32_operand_as_bf16(pa);
    }
#pragma unroll
    for (int j = 0; j < 4; ++j)
      bmat[j] = load_bf16_operand(Vb + (size_t)(16 * j + ln) * SEQ + k0 + kb);
#pragma unroll
    for (int i = 0; i < 4; ++i)
#pragma unroll
      for (int j = 0; j < 4; ++j) acc[i][j] = wmma_bf16(a[i], bmat[j], acc[i][j]);
  }

  const int mrow = (lane >> 4) << 3;
#pragma unroll
  for (int i = 0; i < 4; ++i)
#pragma unroll
    for (int j = 0; j < 4; ++j)
#pragma unroll
      for (int r = 0; r < 8; ++r) {
        const int q = q0 + 16 * i + r + mrow;
        const int d = 16 * j + ln;
        AO[((size_t)b * SEQ + q) * DEMB + h * DHEAD + d] = (bf16_t)acc[i][j][r];
      }
}

// --- Host orchestration ------------------------------------------------------

extern "C" void kernel_launch(void* const* d_in, const int* in_sizes, int n_in,
                              void* d_out, int out_size, void* d_ws, size_t ws_size,
                              hipStream_t stream) {
  (void)in_sizes; (void)n_in; (void)out_size; (void)ws_size;

  const float* x1 = (const float*)d_in[0];
  const float* x2 = (const float*)d_in[1];
  const float* Wq = (const float*)d_in[2];
  const float* bq = (const float*)d_in[3];
  const float* Wk = (const float*)d_in[4];
  const float* bk = (const float*)d_in[5];
  const float* Wv = (const float*)d_in[6];
  const float* bv = (const float*)d_in[7];
  const float* Wo = (const float*)d_in[8];
  const float* bo = (const float*)d_in[9];

  float* out   = (float*)d_out;                       // [B, SEQ, 1024]
  float* attnW = out + (size_t)BATCH * SEQ * DEMB;    // [B, H, SEQ, SEQ]

  // Workspace carve-out (all bf16): ~58.7 MB total.
  bf16_t* ws = (bf16_t*)d_ws;
  const size_t nx = (size_t)BATCH * SEQ * DEMB;   // 4,194,304
  const size_t nw = (size_t)DEMB * DEMB;          // 1,048,576
  bf16_t* x1b = ws; ws += nx;
  bf16_t* x2b = ws; ws += nx;
  bf16_t* wqb = ws; ws += nw;
  bf16_t* wkb = ws; ws += nw;
  bf16_t* wvb = ws; ws += nw;
  bf16_t* wob = ws; ws += nw;
  bf16_t* Qb  = ws; ws += nx;   // [B,H,SEQ,64]
  bf16_t* Kb  = ws; ws += nx;   // [B,H,SEQ,64]
  bf16_t* Vtb = ws; ws += nx;   // [B,H,64,SEQ]
  bf16_t* AOb = ws; ws += nx;   // [B,SEQ,1024]

  // 0) down-convert activations and weights to bf16
  cvt_f32_to_bf16<<<(int)(nx / 256), 256, 0, stream>>>(x1, x1b, (int)nx);
  cvt_f32_to_bf16<<<(int)(nx / 256), 256, 0, stream>>>(x2, x2b, (int)nx);
  cvt_f32_to_bf16<<<(int)(nw / 256), 256, 0, stream>>>(Wq, wqb, (int)nw);
  cvt_f32_to_bf16<<<(int)(nw / 256), 256, 0, stream>>>(Wk, wkb, (int)nw);
  cvt_f32_to_bf16<<<(int)(nw / 256), 256, 0, stream>>>(Wv, wvb, (int)nw);
  cvt_f32_to_bf16<<<(int)(nw / 256), 256, 0, stream>>>(Wo, wob, (int)nw);

  // 1) Q/K/V projections (WMMA); V stored pre-transposed per head
  dim3 gp(MROWS / 128, DEMB / 128);  // (32, 8)
  gemm_xwT_wmma<1><<<gp, 128, 0, stream>>>(x1b, wqb, bq, nullptr, Qb);
  gemm_xwT_wmma<1><<<gp, 128, 0, stream>>>(x2b, wkb, bk, nullptr, Kb);
  gemm_xwT_wmma<2><<<gp, 128, 0, stream>>>(x2b, wvb, bv, nullptr, Vtb);

  // 2) scores = Q K^T / 8 -> attn-weights region of d_out
  dim3 gs(SEQ / 128, SEQ / 128, BATCH * NHEADS);  // (16, 16, 32)
  scores_wmma<<<gs, 128, 0, stream>>>(Qb, Kb, attnW);

  // 3) softmax in place (final attn_weights output)
  softmax_rows<<<BATCH * NHEADS * SEQ, 256, 0, stream>>>(attnW);

  // 4) attn_out = P @ V (WMMA, P converted to bf16 in-register)
  dim3 ga(SEQ / 256, 1, BATCH * NHEADS);  // (8, 1, 32)
  av_wmma<<<ga, 128, 0, stream>>>(attnW, Vtb, AOb);

  // 5) output = attn_out @ Wo^T + bo (f32 out)
  gemm_xwT_wmma<0><<<gp, 128, 0, stream>>>(AOb, wob, bo, out, nullptr);
}
